// LinearFLHWrapper_1176821039710
// MI455X (gfx1250) — 8-bit matrix operand order repaired and checked by reading, compiled, not run on hardware
//
#include <hip/hip_runtime.h>
#include <hip/hip_bf16.h>


#define BDIM 8192
#define KDIM 4096
#define ODIM 16384
#define GRP  128
#define NG   32
#define KB64 (KDIM / 64)

typedef __attribute__((ext_vector_type(8))) int   v8i;
typedef __attribute__((ext_vector_type(4))) int   v4i;
typedef __attribute__((ext_vector_type(8))) float v8f;
typedef __attribute__((ext_vector_type(4))) float v4f;

__device__ __forceinline__ float wave_amax(float a, int lane) {
  #pragma unroll
  for (int m = 16; m >= 1; m >>= 1)
    a = fmaxf(a, __shfl_xor(a, m, 32));
  return a;
}

__global__ __launch_bounds__(256) void act_quant_kernel(
    const float* __restrict__ x, signed char* __restrict__ qxA,
    float* __restrict__ sxT)
{
  const int lane = threadIdx.x & 31;
  const int wave = blockIdx.x * 8 + (threadIdx.x >> 5);
  const int g = wave & (NG - 1);
  const int b = wave >> 5;

  v4f v = *(const v4f*)(x + (size_t)b * KDIM + g * GRP + lane * 4);

  float amax = fmaxf(fmaxf(fabsf(v[0]), fabsf(v[1])),
                     fmaxf(fabsf(v[2]), fabsf(v[3])));
  amax = wave_amax(amax, lane);

  const float scale = fmaxf(amax, 1e-8f) * (1.0f / 7.0f);
  const float inv   = 1.0f / scale;

  int packed = 0;
  #pragma unroll
  for (int j = 0; j < 4; ++j) {
    float qf = fminf(fmaxf(rintf(v[j] * inv), -8.0f), 7.0f);
    packed |= ((int)qf & 0xff) << (8 * j);
  }

  const int kk0 = lane * 4;
  const int kb  = 2 * g + (kk0 >> 6);
  const int Kt  = kk0 & 63;
  const int hi  = (Kt >> 3) & 1;
  const int lt  = (b & 15) + 16 * hi;
  const int vv  = (Kt >> 4) * 2 + ((Kt >> 2) & 1);

  const size_t tile = (size_t)((b >> 4) * KB64 + kb);
  *(int*)(qxA + tile * 1024 + lt * 32 + vv * 4) = packed;
  if (lane == 0) sxT[(size_t)g * BDIM + b] = scale;
}

__global__ __launch_bounds__(256) void wt_quant_kernel(
    const float* __restrict__ w, signed char* __restrict__ qwB,
    float* __restrict__ swT)
{
  const int lane = threadIdx.x & 31;
  const int wave = blockIdx.x * 8 + (threadIdx.x >> 5);
  const int g = wave & (NG - 1);
  const int o = wave >> 5;

  v4f v = *(const v4f*)(w + (size_t)o * KDIM + g * GRP + lane * 4);

  float a0 = v[0], a1 = v[1], a2 = v[2], a3 = v[3], t;
  t = a0; a0 = t + a1; a1 = t - a1;
  t = a2; a2 = t + a3; a3 = t - a3;
  t = a0; a0 = t + a2; a2 = t - a2;
  t = a1; a1 = t + a3; a3 = t - a3;
  float arr[4] = {a0, a1, a2, a3};
  #pragma unroll
  for (int m = 1; m <= 16; m <<= 1) {
    #pragma unroll
    for (int j = 0; j < 4; ++j) {
      float other = __shfl_xor(arr[j], m, 32);
      arr[j] = (lane & m) ? (other - arr[j]) : (arr[j] + other);
    }
  }
  const float norm = 0.08838834764831845f;
  float amax = 0.0f;
  #pragma unroll
  for (int j = 0; j < 4; ++j) { arr[j] *= norm; amax = fmaxf(amax, fabsf(arr[j])); }
  amax = wave_amax(amax, lane);

  const float scale = fmaxf(amax, 1e-8f) * (1.0f / 7.0f);
  const float inv   = 1.0f / scale;

  int packed = 0;
  #pragma unroll
  for (int j = 0; j < 4; ++j) {
    float qf = fminf(fmaxf(rintf(arr[j] * inv), -8.0f), 7.0f);
    packed |= ((int)qf & 0xff) << (8 * j);
  }

  const int kk0 = lane * 4;
  const int kb  = 2 * g + (kk0 >> 6);
  const int Kt  = kk0 & 63;
  const int lt  = ((Kt >> 3) & 1) * 16 + (o & 15);
  const int vv  = (Kt >> 4) * 2 + ((Kt >> 2) & 1);

  const size_t tile = (size_t)((o >> 4) * KB64 + kb);
  *(int*)(qwB + tile * 1024 + lt * 32 + vv * 4) = packed;
  if (lane == 0) swT[(size_t)g * ODIM + o] = scale;
}

__global__ __launch_bounds__(256) void i4_gemm_kernel(
    const signed char* __restrict__ qxA, const signed char* __restrict__ qwB,
    const float* __restrict__ sxT, const float* __restrict__ swT,
    float* __restrict__ out)
{
  const int lane = threadIdx.x & 31;
  const int wv   = threadIdx.x >> 5;
  const int wm   = wv >> 2, wn = wv & 3;
  const int mt0  = blockIdx.y * 4 + wm * 2;
  const int nt0  = blockIdx.x * 8 + wn * 2;
  const int hi   = lane >> 4;
  const int ln   = lane & 15;

  v8f acc[2][2];
  #pragma unroll
  for (int mi = 0; mi < 2; ++mi)
    #pragma unroll
    for (int ni = 0; ni < 2; ++ni)
      #pragma unroll
      for (int i = 0; i < 8; ++i) acc[mi][ni][i] = 0.0f;

  const signed char* aBase[2];
  const signed char* bBase[2];
  #pragma unroll
  for (int mi = 0; mi < 2; ++mi)
    aBase[mi] = qxA + (size_t)(mt0 + mi) * KB64 * 1024 + lane * 32;
  #pragma unroll
  for (int ni = 0; ni < 2; ++ni)
    bBase[ni] = qwB + (size_t)(nt0 + ni) * KB64 * 1024 + lane * 32;

  for (int g = 0; g < NG; ++g) {
    if (g + 1 < NG) {
      __builtin_prefetch(aBase[0] + (size_t)(2 * g + 2) * 1024, 0, 3);
      __builtin_prefetch(bBase[0] + (size_t)(2 * g + 2) * 1024, 0, 3);
    }

    v8f sxv[2];
    #pragma unroll
    for (int mi = 0; mi < 2; ++mi) {
      const float* p = sxT + (size_t)g * BDIM + (mt0 + mi) * 16 + hi * 8;
      v4f lo = *(const v4f*)p;
      v4f hb = *(const v4f*)(p + 4);
      #pragma unroll
      for (int i = 0; i < 4; ++i) { sxv[mi][i] = lo[i]; sxv[mi][i + 4] = hb[i]; }
    }
    float swv[2];
    #pragma unroll
    for (int ni = 0; ni < 2; ++ni)
      swv[ni] = swT[(size_t)g * ODIM + (nt0 + ni) * 16 + ln];

    v8i ia[2][2];
    #pragma unroll
    for (int mi = 0; mi < 2; ++mi)
      #pragma unroll
      for (int ni = 0; ni < 2; ++ni)
        #pragma unroll
        for (int i = 0; i < 8; ++i) ia[mi][ni][i] = 0;

    #pragma unroll
    for (int ks = 0; ks < 2; ++ks) {
      const size_t kb = (size_t)(2 * g + ks) * 1024;
      v8i af[2], bf[2];
      #pragma unroll
      for (int mi = 0; mi < 2; ++mi) {
        v4i lo = *(const v4i*)(aBase[mi] + kb);
        v4i hb = *(const v4i*)(aBase[mi] + kb + 16);
        #pragma unroll
        for (int i = 0; i < 4; ++i) { af[mi][i] = lo[i]; af[mi][i + 4] = hb[i]; }
      }
      #pragma unroll
      for (int ni = 0; ni < 2; ++ni) {
        v4i lo = *(const v4i*)(bBase[ni] + kb);
        v4i hb = *(const v4i*)(bBase[ni] + kb + 16);
        #pragma unroll
        for (int i = 0; i < 4; ++i) { bf[ni][i] = lo[i]; bf[ni][i + 4] = hb[i]; }
      }
      #pragma unroll
      for (int mi = 0; mi < 2; ++mi)
        #pragma unroll
        for (int ni = 0; ni < 2; ++ni)
          ia[mi][ni] = __builtin_amdgcn_wmma_i32_16x16x64_iu8(
               true, af[mi],  true, bf[ni],
              ia[mi][ni],  false,  false);
    }

    #pragma unroll
    for (int mi = 0; mi < 2; ++mi)
      #pragma unroll
      for (int ni = 0; ni < 2; ++ni)
        #pragma unroll
        for (int i = 0; i < 8; ++i)
          acc[mi][ni][i] += (float)ia[mi][ni][i] * (sxv[mi][i] * swv[ni]);
  }

  #pragma unroll
  for (int mi = 0; mi < 2; ++mi)
    #pragma unroll
    for (int ni = 0; ni < 2; ++ni) {
      const int N = (nt0 + ni) * 16 + ln;
      #pragma unroll
      for (int i = 0; i < 8; ++i) {
        const int M = (mt0 + mi) * 16 + i + 8 * hi;
        out[(size_t)M * ODIM + N] = acc[mi][ni][i];
      }
    }
}

extern "C" void kernel_launch(void* const* d_in, const int* in_sizes, int n_in,
                              void* d_out, int out_size, void* d_ws, size_t ws_size,
                              hipStream_t stream) {
  const float* x = (const float*)d_in[0];
  const float* w = (const float*)d_in[1];
  float* out = (float*)d_out;

  char* ws = (char*)d_ws;
  signed char* qxA = (signed char*)ws;
  signed char* qwB = (signed char*)(ws + (size_t)BDIM * KDIM);
  float* sxT = (float*)(ws + (size_t)BDIM * KDIM + (size_t)ODIM * KDIM);
  float* swT = sxT + (size_t)NG * BDIM;

  act_quant_kernel<<<(BDIM * NG) / 8, 256, 0, stream>>>(x, qxA, sxT);
  wt_quant_kernel<<<(ODIM * NG) / 8, 256, 0, stream>>>(w, qwB, swT);

  dim3 grid(ODIM / 128, BDIM / 64);
  i4_gemm_kernel<<<grid, 256, 0, stream>>>(qxA, qwB, sxT, swT, out);
}
